// ChebNN_9345848836279
// MI455X (gfx1250) — matrix operand order repaired and checked by reading, not compiled, not run on hardware
//
#include <hip/hip_runtime.h>
#include <hip/hip_bf16.h>
#include <cmath>

// ---------------- problem constants (match reference) ----------------
constexpr int NNODES = 50000;   // 3125 * 16 = 1562*32 + 16
constexpr int NEDGES = 800000;
constexpr int NFEAT  = 512;
constexpr int HIDDIM = 256;
constexpr int NCLS   = 64;
constexpr int KSTEPS = 10;      // scan runs K+1 = 11 steps

// ---------------- WMMA types ----------------
typedef __attribute__((ext_vector_type(16))) __bf16 v16bf;
typedef __attribute__((ext_vector_type(8)))  __bf16 v8bf;
typedef __attribute__((ext_vector_type(8)))  float  v8f;
typedef __attribute__((ext_vector_type(4)))  __bf16 v4bf;

// ---------------- gfx1250 async global->LDS copy (ASYNCcnt) ----------------
__device__ __forceinline__ void async_ld16(unsigned lds_addr, unsigned long long gaddr) {
    asm volatile("global_load_async_to_lds_b128 %0, %1, off"
                 :: "v"(lds_addr), "v"(gaddr) : "memory");
}
__device__ __forceinline__ void async_wait0() {
    asm volatile("s_wait_asynccnt 0x0" ::: "memory");
}

// Stage a contiguous slab (16 rows x K bf16, rows contiguous) into LDS.
// BYTES = bytes per thread; must be a multiple of 16.
template <int BYTES>
__device__ __forceinline__ void stage_slab(const __bf16* g, __bf16* lds, int tid) {
    unsigned long long ga = (unsigned long long)(uintptr_t)g + (unsigned)tid * BYTES;
    // low 32 bits of a generic shared pointer == LDS byte offset (ISA aperture rule)
    unsigned la = (unsigned)(uintptr_t)lds + (unsigned)tid * BYTES;
#pragma unroll
    for (int o = 0; o < BYTES; o += 16) async_ld16(la + o, ga + o);
}

// Concatenate two 8-element chunks into a 16-element fragment.
__device__ __forceinline__ v16bf cat8(v8bf lo, v8bf hi) {
    return __builtin_shufflevector(lo, hi, 0, 1, 2, 3, 4, 5, 6, 7,
                                   8, 9, 10, 11, 12, 13, 14, 15);
}

// ---------------- fragment loaders ----------------
// A fragment (16x32 bf16) from an LDS slab [16][lda], rows 0..15.
// Layout: lanes 0-15: M=lane, elems 0..15 -> K=k0+0..15;
// lanes 16-31: same with K offset +16.
// Two native 16B vector loads -> two ds_load_b128.
__device__ __forceinline__ v16bf frag_a_slab(const __bf16* As, int lda, int k0, int lane) {
    const __bf16* p = As + (size_t)(lane & 15) * lda + k0 + ((lane >> 4) << 4);
    v8bf lo = *(const v8bf*)p;
    v8bf hi = *(const v8bf*)(p + 8);
    return cat8(lo, hi);
}

// B fragment (32x16) from bf16 N-major transposed weights Wt[n][k] (ldk = K).
// Lane holds column n = col0+(lane&15); 16 contiguous K values (32B-aligned)
// -> exactly two global_load_b128 per lane.
__device__ __forceinline__ v16bf frag_b_t(const __bf16* Wt, int ldk, int k0, int col0, int lane) {
    const __bf16* p = Wt + (size_t)(col0 + (lane & 15)) * ldk + k0 + ((lane >> 4) << 4);
    v8bf lo = *(const v8bf*)p;
    v8bf hi = *(const v8bf*)(p + 8);
    return cat8(lo, hi);
}

// GEMM mainloop: NSLAB row slabs (16 rows each, LDS) x 2 column tiles.
// B fragments are loaded once per k-chunk and reused by every row slab.
template <int NSLAB, int KDIM>
__device__ __forceinline__ void gemm_slabs(const __bf16* As, const __bf16* __restrict__ Wt,
                                           int n1, int n2, int lane, v8f (&acc)[NSLAB][2]) {
#pragma unroll
    for (int k0 = 0; k0 < KDIM; k0 += 32) {
        v16bf b1 = frag_b_t(Wt, KDIM, k0, n1, lane);
        v16bf b2 = frag_b_t(Wt, KDIM, k0, n2, lane);
#pragma unroll
        for (int s = 0; s < NSLAB; ++s) {
            v16bf a = frag_a_slab(As + (size_t)s * 16 * KDIM, KDIM, k0, lane);
            acc[s][0] = __builtin_amdgcn_wmma_f32_16x16x32_bf16(false, a, false, b1,
                                                               (short)0, acc[s][0], false, false);
            acc[s][1] = __builtin_amdgcn_wmma_f32_16x16x32_bf16(false, a, false, b2,
                                                               (short)0, acc[s][1], false, false);
        }
    }
}

// ---------------- prep: f32 -> bf16 convert (contiguous) ----------------
__global__ __launch_bounds__(256) void cvt_bf16_kernel(const float* __restrict__ src,
                                                       __bf16* __restrict__ dst, int n4) {
    int i = (int)(blockIdx.x * blockDim.x + threadIdx.x);
    if (i >= n4) return;
    float4 x = ((const float4*)src)[i];
    v4bf q;
    q[0] = (__bf16)x.x; q[1] = (__bf16)x.y; q[2] = (__bf16)x.z; q[3] = (__bf16)x.w;
    ((v4bf*)dst)[i] = q;
}

// ---------------- prep: transpose + convert: dst[n*K+k] = (bf16)src[k*N+n] --
__global__ __launch_bounds__(256) void transpose_cvt_kernel(const float* __restrict__ src,
                                                            __bf16* __restrict__ dst,
                                                            int Kd, int Nd) {
    int idx = (int)(blockIdx.x * blockDim.x + threadIdx.x);
    if (idx >= Kd * Nd) return;
    int n = idx / Kd, k = idx - n * Kd;
    dst[idx] = (__bf16)src[(size_t)k * Nd + n];
}

// ---------------- fc1: h0 = relu(X @ W1 + b1), M=50000 K=512 N=256 ---------
__device__ __forceinline__ void fc1_epilogue(int rbase, int n1, int n2, int lane,
                                             v8f (&acc)[2], const float* __restrict__ bias,
                                             float* __restrict__ H0) {
    const int m0 = rbase + ((lane >> 4) << 3);
    const int c1 = n1 + (lane & 15), c2 = n2 + (lane & 15);
    const float b1v = bias[c1], b2v = bias[c2];
#pragma unroll
    for (int v = 0; v < 8; ++v) {
        H0[(size_t)(m0 + v) * HIDDIM + c1] = fmaxf(acc[0][v] + b1v, 0.0f);
        H0[(size_t)(m0 + v) * HIDDIM + c2] = fmaxf(acc[1][v] + b2v, 0.0f);
    }
}

__global__ __launch_bounds__(256) void fc1_kernel(const __bf16* __restrict__ Xbf,
                                                  const __bf16* __restrict__ Wt,  // [256][512]
                                                  const float* __restrict__ bias,
                                                  float* __restrict__ H0) {
    __shared__ __bf16 As[32 * NFEAT];                 // 32 KB
    const int tid  = (int)threadIdx.x;
    const int row0 = (int)blockIdx.x * 32;
    const bool two = (row0 + 32) <= NNODES;           // last block covers only 16 rows
    stage_slab<64>(Xbf + (size_t)row0 * NFEAT, As, tid);
    if (two) stage_slab<64>(Xbf + (size_t)(row0 + 16) * NFEAT, As + 16 * NFEAT, tid);
    async_wait0();
    __syncthreads();

    const int lane = tid & 31, wid = tid >> 5;
    const int n1 = wid * 16, n2 = (wid + 8) * 16;     // two col tiles per wave
    if (two) {
        v8f acc[2][2] = {};
        gemm_slabs<2, NFEAT>(As, Wt, n1, n2, lane, acc);
        fc1_epilogue(row0,      n1, n2, lane, acc[0], bias, H0);
        fc1_epilogue(row0 + 16, n1, n2, lane, acc[1], bias, H0);
    } else {
        v8f acc[1][2] = {};
        gemm_slabs<1, NFEAT>(As, Wt, n1, n2, lane, acc);
        fc1_epilogue(row0, n1, n2, lane, acc[0], bias, H0);
    }
}

// ---------------- zero a buffer (float4) ----------------
__global__ __launch_bounds__(256) void zero_kernel(float4* __restrict__ p, int n4) {
    int i = (int)(blockIdx.x * blockDim.x + threadIdx.x);
    if (i < n4) p[i] = make_float4(0.0f, 0.0f, 0.0f, 0.0f);
}

// ---------------- scatter: agg[dst] += norm * h[src], HW f32 atomics -------
__global__ __launch_bounds__(256) void scatter_kernel(const float* __restrict__ H,
                                                      const int* __restrict__ srcIdx,
                                                      const int* __restrict__ dstIdx,
                                                      const float* __restrict__ norm,
                                                      float* __restrict__ agg) {
    const int e    = (int)((blockIdx.x * blockDim.x + threadIdx.x) >> 5);
    const int lane = (int)(threadIdx.x & 31);
    if (e >= NEDGES) return;                          // wave-uniform guard
    const int   s = srcIdx[e];
    const int   d = dstIdx[e];
    const float w = norm[e];
    const float4* hp = (const float4*)(H + (size_t)s * HIDDIM) + lane * 2;
    float*        ap = agg + (size_t)d * HIDDIM + lane * 8;
    const float4 x0 = hp[0];
    const float4 x1 = hp[1];
    unsafeAtomicAdd(ap + 0, w * x0.x);
    unsafeAtomicAdd(ap + 1, w * x0.y);
    unsafeAtomicAdd(ap + 2, w * x0.z);
    unsafeAtomicAdd(ap + 3, w * x0.w);
    unsafeAtomicAdd(ap + 4, w * x1.x);
    unsafeAtomicAdd(ap + 5, w * x1.y);
    unsafeAtomicAdd(ap + 6, w * x1.z);
    unsafeAtomicAdd(ap + 7, w * x1.w);
}

// ---------------- combine: t = 2*agg - h_prev2 + alpha[aidx]*h0 ------------
__global__ __launch_bounds__(256) void combine_kernel(float* __restrict__ t,
                                                      const float* __restrict__ hprev2,
                                                      const float* __restrict__ h0,
                                                      const float* __restrict__ alpha,
                                                      int aidx,
                                                      __bf16* __restrict__ tbf,
                                                      int n4) {
    int i = (int)(blockIdx.x * blockDim.x + threadIdx.x);
    if (i >= n4) return;
    const float a = alpha[aidx];
    float4 g = ((const float4*)t)[i];
    float4 p = ((const float4*)hprev2)[i];
    float4 z = ((const float4*)h0)[i];
    float4 r;
    r.x = 2.0f * g.x - p.x + a * z.x;
    r.y = 2.0f * g.y - p.y + a * z.y;
    r.z = 2.0f * g.z - p.z + a * z.z;
    r.w = 2.0f * g.w - p.w + a * z.w;
    ((float4*)t)[i] = r;
    v4bf q;
    q[0] = (__bf16)r.x; q[1] = (__bf16)r.y; q[2] = (__bf16)r.z; q[3] = (__bf16)r.w;
    ((v4bf*)tbf)[i] = q;
}

// ---------------- conv step: h = (1-b)*t + b*(t @ Wk + bk) -----------------
__device__ __forceinline__ void conv_epilogue(int rbase, int n1, int n2, int lane,
                                              v8f (&acc)[2],
                                              const float* __restrict__ T,
                                              const float* __restrict__ bias,
                                              float beta, int do_relu, int emit_bf,
                                              float* __restrict__ Hout,
                                              __bf16* __restrict__ Hbf) {
    const int m0 = rbase + ((lane >> 4) << 3);
    const int c1 = n1 + (lane & 15), c2 = n2 + (lane & 15);
    const float b1v = bias[c1], b2v = bias[c2];
    const float onemb = 1.0f - beta;
#pragma unroll
    for (int v = 0; v < 8; ++v) {
        const size_t i1 = (size_t)(m0 + v) * HIDDIM + c1;
        const size_t i2 = (size_t)(m0 + v) * HIDDIM + c2;
        float h1 = onemb * T[i1] + beta * (acc[0][v] + b1v);
        float h2 = onemb * T[i2] + beta * (acc[1][v] + b2v);
        if (do_relu) { h1 = fmaxf(h1, 0.0f); h2 = fmaxf(h2, 0.0f); }
        Hout[i1] = h1;
        Hout[i2] = h2;
        if (emit_bf) {                                 // fc2 input: relu(last_h) in bf16
            Hbf[i1] = (__bf16)fmaxf(h1, 0.0f);
            Hbf[i2] = (__bf16)fmaxf(h2, 0.0f);
        }
    }
}

__global__ __launch_bounds__(256) void conv_kernel(const __bf16* __restrict__ Tbf,
                                                   const float* __restrict__ T,
                                                   const __bf16* __restrict__ Wt,  // [256][256]
                                                   const float* __restrict__ bias,
                                                   float beta, int do_relu, int emit_bf,
                                                   float* __restrict__ Hout,
                                                   __bf16* __restrict__ Hbf) {
    __shared__ __bf16 As[32 * HIDDIM];                // 16 KB
    const int tid  = (int)threadIdx.x;
    const int row0 = (int)blockIdx.x * 32;
    const bool two = (row0 + 32) <= NNODES;
    stage_slab<32>(Tbf + (size_t)row0 * HIDDIM, As, tid);
    if (two) stage_slab<32>(Tbf + (size_t)(row0 + 16) * HIDDIM, As + 16 * HIDDIM, tid);
    async_wait0();
    __syncthreads();

    const int lane = tid & 31, wid = tid >> 5;
    const int n1 = wid * 16, n2 = (wid + 8) * 16;
    if (two) {
        v8f acc[2][2] = {};
        gemm_slabs<2, HIDDIM>(As, Wt, n1, n2, lane, acc);
        conv_epilogue(row0,      n1, n2, lane, acc[0], T, bias, beta, do_relu, emit_bf, Hout, Hbf);
        conv_epilogue(row0 + 16, n1, n2, lane, acc[1], T, bias, beta, do_relu, emit_bf, Hout, Hbf);
    } else {
        v8f acc[1][2] = {};
        gemm_slabs<1, HIDDIM>(As, Wt, n1, n2, lane, acc);
        conv_epilogue(row0, n1, n2, lane, acc[0], T, bias, beta, do_relu, emit_bf, Hout, Hbf);
    }
}

// ---------------- fc2: out = A @ W2 + b2 (A = relu'd bf16), N=64 -----------
__global__ __launch_bounds__(128) void fc2_kernel(const __bf16* __restrict__ Abf,
                                                  const __bf16* __restrict__ Wt,  // [64][256]
                                                  const float* __restrict__ bias,
                                                  float* __restrict__ Out) {
    __shared__ __bf16 As[16 * HIDDIM];                // 8 KB
    const int tid  = (int)threadIdx.x;
    const int row0 = (int)blockIdx.x * 16;
    stage_slab<64>(Abf + (size_t)row0 * HIDDIM, As, tid);  // 128 thr * 64B = 8KB
    async_wait0();
    __syncthreads();

    const int lane = tid & 31, wid = tid >> 5;        // 4 waves -> 4 col tiles
    const int n1 = wid * 16;
    v8f acc = {};
#pragma unroll
    for (int k0 = 0; k0 < HIDDIM; k0 += 32) {
        v16bf a = frag_a_slab(As, HIDDIM, k0, lane);
        v16bf b = frag_b_t(Wt, HIDDIM, k0, n1, lane);
        acc = __builtin_amdgcn_wmma_f32_16x16x32_bf16(false, a, false, b, (short)0, acc, false, false);
    }
    const int m0 = row0 + ((lane >> 4) << 3);
    const int c  = n1 + (lane & 15);
    const float bn = bias[c];
#pragma unroll
    for (int v = 0; v < 8; ++v)
        Out[(size_t)(m0 + v) * NCLS + c] = acc[v] + bn;
}

// ---------------- host orchestration ---------------------------------------
extern "C" void kernel_launch(void* const* d_in, const int* in_sizes, int n_in,
                              void* d_out, int out_size, void* d_ws, size_t ws_size,
                              hipStream_t stream) {
    const float* features = (const float*)d_in[0];
    const int*   eidx     = (const int*)d_in[1];   // [2, E]: row0 = src, row1 = dst
    const float* normA    = (const float*)d_in[2];
    const float* convW    = (const float*)d_in[3]; // [K+1, 256, 256]
    const float* convB    = (const float*)d_in[4];
    const float* fc1W     = (const float*)d_in[5]; // [512, 256]
    const float* fc1b     = (const float*)d_in[6];
    const float* fc2W     = (const float*)d_in[7]; // [256, 64]
    const float* fc2b     = (const float*)d_in[8];
    const float* alpha    = (const float*)d_in[9];

    const int* srcIdx = eidx;
    const int* dstIdx = eidx + NEDGES;

    const size_t nh = (size_t)NNODES * HIDDIM;     // 12.8M
    const size_t nf = (size_t)NNODES * NFEAT;      // 25.6M
    float*  h0     = (float*)d_ws;
    float*  hA     = h0 + nh;
    float*  hB     = hA + nh;
    float*  agg    = hB + nh;                      // doubles as t after combine
    __bf16* tbf    = (__bf16*)(agg + nh);          // bf16(t); last step: relu(h) bf16
    __bf16* featbf = tbf + nh;
    __bf16* wtconv = featbf + nf;                  // 11 x [256][256]
    __bf16* wtfc1  = wtconv + (size_t)(KSTEPS + 1) * HIDDIM * HIDDIM;  // [256][512]
    __bf16* wtfc2  = wtfc1 + (size_t)HIDDIM * NFEAT;                   // [64][256]

    const dim3 blk(256);
    const int slab32_blocks = (NNODES + 31) / 32;                   // 1563
    const int n4          = (int)(nh / 4);
    const int elem_blocks = (n4 + 255) / 256;                       // 12500
    const int scat_blocks = NEDGES / 8;                             // 100000

    // ---- prep: bf16 conversions / weight transposes (small, once per call)
    cvt_bf16_kernel<<<(int)(nf / 4 + 255) / 256, blk, 0, stream>>>(features, featbf, (int)(nf / 4));
    for (int i = 0; i <= KSTEPS; ++i)
        transpose_cvt_kernel<<<(HIDDIM * HIDDIM + 255) / 256, blk, 0, stream>>>(
            convW + (size_t)i * HIDDIM * HIDDIM, wtconv + (size_t)i * HIDDIM * HIDDIM,
            HIDDIM, HIDDIM);
    transpose_cvt_kernel<<<(NFEAT * HIDDIM + 255) / 256, blk, 0, stream>>>(fc1W, wtfc1, NFEAT, HIDDIM);
    transpose_cvt_kernel<<<(HIDDIM * NCLS + 255) / 256, blk, 0, stream>>>(fc2W, wtfc2, HIDDIM, NCLS);

    // ---- h0 = relu(X @ W1 + b1)
    fc1_kernel<<<slab32_blocks, blk, 0, stream>>>(featbf, wtfc1, fc1b, h0);

    // ---- scan carry init: last_h = 0, second_last_h = 0
    zero_kernel<<<elem_blocks, blk, 0, stream>>>((float4*)hA, n4);
    zero_kernel<<<elem_blocks, blk, 0, stream>>>((float4*)hB, n4);

    float* hL = hA;   // last_h
    float* hP = hB;   // second_last_h
    for (int i = 0; i <= KSTEPS; ++i) {
        zero_kernel<<<elem_blocks, blk, 0, stream>>>((float4*)agg, n4);
        scatter_kernel<<<scat_blocks, blk, 0, stream>>>(hL, srcIdx, dstIdx, normA, agg);
        // t = 2*agg - second_last + alpha[K-i]*h0
        combine_kernel<<<elem_blocks, blk, 0, stream>>>(agg, hP, h0, alpha,
                                                        KSTEPS - i, tbf, n4);
        const float beta = logf(1.0f / (float)(i + 1) + 1.0f);
        conv_kernel<<<slab32_blocks, blk, 0, stream>>>(
            tbf, agg, wtconv + (size_t)i * HIDDIM * HIDDIM, convB + (size_t)i * HIDDIM,
            beta, (i < KSTEPS - 1) ? 1 : 0, (i == KSTEPS) ? 1 : 0, hP, tbf);
        float* tmp = hL; hL = hP; hP = tmp;
    }

    // ---- out = relu(last_h) @ W2 + b2  (A already relu'd bf16 in tbf)
    fc2_kernel<<<NNODES / 16, dim3(128), 0, stream>>>(tbf, wtfc2, fc2b, (float*)d_out);
}
